// GraphAttentionLayer_55688545960644
// MI455X (gfx1250) — compile-verified, matrix operand order checked by reading, not run on hardware
//
#include <hip/hip_runtime.h>
#include <hip/hip_bf16.h>

#define GAT_N 8192
#define GAT_D 256
#define NEG_BIG (-9.0e15f)

typedef __attribute__((ext_vector_type(16))) _Float16 v16h;
typedef __attribute__((ext_vector_type(8)))  _Float16 v8h;
typedef __attribute__((ext_vector_type(4)))  _Float16 v4h;
typedef __attribute__((ext_vector_type(8)))  float    v8f;
typedef __attribute__((ext_vector_type(4)))  float    v4f;
typedef __attribute__((ext_vector_type(4)))  int      v4i;

// ---------------------------------------------------------------------------
// w1 = W^T @ a1, w2 = W^T @ a2, cb = {bW.a1, bW.a2}.  (f1 = X@w1 + cb0, etc.)
// One block; trivial cost.
// ---------------------------------------------------------------------------
__global__ __launch_bounds__(256)
void gat_w12(const float* __restrict__ W, const float* __restrict__ bW,
             const float* __restrict__ a_vec, float* __restrict__ w1,
             float* __restrict__ w2, float* __restrict__ cb) {
  const int k = threadIdx.x;           // 0..255 (column of W)
  float s1 = 0.f, s2 = 0.f;
  for (int n = 0; n < GAT_D; ++n) {
    const float w = W[(size_t)n * GAT_D + k];
    s1 += a_vec[n] * w;
    s2 += a_vec[GAT_D + n] * w;
  }
  w1[k] = s1;
  w2[k] = s2;
  if (k == 0) {
    float c1 = 0.f, c2 = 0.f;
    for (int n = 0; n < GAT_D; ++n) {
      c1 += bW[n] * a_vec[n];
      c2 += bW[n] * a_vec[GAT_D + n];
    }
    cb[0] = c1;
    cb[1] = c2;
  }
}

// ---------------------------------------------------------------------------
// f1[i] = X[i,:].w1 + cb0 ; f2[i] = X[i,:].w2 + cb1.  Coalesced-per-row GEMV.
// ---------------------------------------------------------------------------
__global__ __launch_bounds__(256)
void gat_f12(const float* __restrict__ X, const float* __restrict__ w1,
             const float* __restrict__ w2, const float* __restrict__ cb,
             float* __restrict__ f1, float* __restrict__ f2) {
  const int i = blockIdx.x * blockDim.x + threadIdx.x;   // row
  const float* xr = X + (size_t)i * GAT_D;
  float s1 = 0.f, s2 = 0.f;
  for (int k = 0; k < GAT_D; k += 4) {
    const v4f x = *(const v4f*)(xr + k);
    const v4f u = *(const v4f*)(w1 + k);
    const v4f v = *(const v4f*)(w2 + k);
    for (int t = 0; t < 4; ++t) { s1 += x[t] * u[t]; s2 += x[t] * v[t]; }
  }
  f1[i] = s1 + cb[0];
  f2[i] = s2 + cb[1];
}

// ---------------------------------------------------------------------------
// Phase 1: h = X @ W^T + bW via v_wmma_f32_16x16x32_f16 (one wave = one
// 16x16 tile). Stores h transposed as f16 (h_t[n][j]) for phase 2's B-operand.
// ---------------------------------------------------------------------------
__global__ __launch_bounds__(32)
void gat_phase1(const float* __restrict__ X, const float* __restrict__ W,
                const float* __restrict__ bW, _Float16* __restrict__ h_t) {
  const int lane  = threadIdx.x;
  const int mtile = blockIdx.x;
  const int ntile = blockIdx.y;
  const int l15   = lane & 15;
  const int half  = lane >> 4;          // 0: K groups {0..7,16..23}; 1: {8..15,24..31}
  const int kb    = half ? 8 : 0;
  const int n     = ntile * 16 + l15;

  const float* xrow = X + (size_t)(mtile * 16 + l15) * GAT_D;
  const float* wrow = W + (size_t)n * GAT_D;   // B[k][n] = W^T[k][n] = W[n][k]

  v8f acc = {};
  for (int k0 = 0; k0 < GAT_D; k0 += 32) {
    v4f xa0 = *(const v4f*)(xrow + k0 + kb);
    v4f xa1 = *(const v4f*)(xrow + k0 + kb + 4);
    v4f xb0 = *(const v4f*)(xrow + k0 + 16 + kb);
    v4f xb1 = *(const v4f*)(xrow + k0 + 16 + kb + 4);
    v4f wa0 = *(const v4f*)(wrow + k0 + kb);
    v4f wa1 = *(const v4f*)(wrow + k0 + kb + 4);
    v4f wb0 = *(const v4f*)(wrow + k0 + 16 + kb);
    v4f wb1 = *(const v4f*)(wrow + k0 + 16 + kb + 4);
    v16h a, b;
    for (int i = 0; i < 4; ++i) {
      a[i]      = (_Float16)xa0[i];  a[4 + i]  = (_Float16)xa1[i];
      a[8 + i]  = (_Float16)xb0[i];  a[12 + i] = (_Float16)xb1[i];
      b[i]      = (_Float16)wa0[i];  b[4 + i]  = (_Float16)wa1[i];
      b[8 + i]  = (_Float16)wb0[i];  b[12 + i] = (_Float16)wb1[i];
    }
    acc = __builtin_amdgcn_wmma_f32_16x16x32_f16(false, a, false, b,
                                                 (short)0, acc, false, false);
  }

  const float bwv = bW[n];
  for (int r = 0; r < 8; ++r) {
    const int mg = mtile * 16 + r + (half ? 8 : 0);
    h_t[(size_t)n * GAT_N + mg] = (_Float16)(acc[r] + bwv);  // transposed f16
  }
}

// ---------------------------------------------------------------------------
// gmax = max_j f2[j]. Softmax is shift-invariant, and e is monotonic in f2,
// so M_i = leaky(f1_i + b_a + gmax) upper-bounds every score in row i.
// ---------------------------------------------------------------------------
__global__ __launch_bounds__(256)
void gat_gmax(const float* __restrict__ f2, float* __restrict__ gmax) {
  __shared__ float red[8];
  float m = -3.0e38f;
  for (int i = threadIdx.x; i < GAT_N; i += 256) m = fmaxf(m, f2[i]);
  for (int off = 16; off >= 1; off >>= 1) m = fmaxf(m, __shfl_xor(m, off, 32));
  if ((threadIdx.x & 31) == 0) red[threadIdx.x >> 5] = m;
  __syncthreads();
  if (threadIdx.x == 0) {
    float g = red[0];
    for (int w = 1; w < 8; ++w) g = fmaxf(g, red[w]);
    gmax[0] = g;
  }
}

// ---------------------------------------------------------------------------
// Phase 2: fused masked-softmax attention with a FIXED per-row reference max
// (no online rescale). Block = 32 rows, 8 waves.
// Softmax mapping: thread t -> row t>>3, cols 4*(t&7)..+3  => one b128
// adjacency load, one b64 p-store per thread per step.
// WMMA mapping: wave w -> N-tiles {2w,2w+1} x both 16-row tiles (4 WMMAs/step).
// Single barrier per step via double-buffered p tile; per-lane running
// denominators reduced once at the end.
// ---------------------------------------------------------------------------
__global__ __launch_bounds__(256)
void gat_phase2(const int* __restrict__ adj, const float* __restrict__ f1,
                const float* __restrict__ f2, const float* __restrict__ b_a,
                const float* __restrict__ gmax, const _Float16* __restrict__ h_t,
                float* __restrict__ out) {
  __shared__ float l_s[32];
  __shared__ __attribute__((aligned(16))) _Float16 ptile[2][2][16][32]; // [buf][rt][m][col]

  const int tid  = threadIdx.x;
  const int lane = tid & 31;
  const int wave = tid >> 5;
  const int l15  = lane & 15;
  const int half = lane >> 4;
  const int kb   = half ? 8 : 0;
  const int row0 = blockIdx.x * 32;

  // softmax-side mapping: one row, four consecutive columns per thread
  const int r  = tid >> 3;             // 0..31 (row within tile)
  const int c0 = (tid & 7) * 4;        // 0,4,...,28 (first column)
  const int rtp = r >> 4;              // row-tile for p stores
  const int mp  = r & 15;              // row within row-tile

  const float ba = b_a[0];
  const float g  = gmax[0];
  const float f1r = f1[row0 + r] + ba;
  const float vM  = f1r + g;
  const float Mr  = vM > 0.f ? vM : 0.2f * vM;   // row-wise score upper bound
  float sl = 0.f;                                 // per-lane partial denominator

  const int nA = (2 * wave) * 16 + l15;
  const int nB = (2 * wave + 1) * 16 + l15;
  const _Float16* hA = h_t + (size_t)nA * GAT_N;
  const _Float16* hB = h_t + (size_t)nB * GAT_N;
  const int* arow = adj + (size_t)(row0 + r) * GAT_N;

  v8f acc[2][2] = {};
  int pb = 0;

  for (int j0 = 0; j0 < GAT_N; j0 += 32) {
    // ---- stage 1: masked scores -> p (one b128 adj load, one b64 p store) ----
    const v4i av  = *(const v4i*)(arow + j0 + c0);
    if (j0 + 32 < GAT_N) __builtin_prefetch(arow + j0 + 32 + c0, 0, 1);
    const v4f f2v = *(const v4f*)(f2 + j0 + c0);
    v4h ph;
    for (int t = 0; t < 4; ++t) {
      float e = f1r + f2v[t];
      e = e > 0.f ? e : 0.2f * e;        // LeakyReLU(0.2)
      e = av[t] > 0 ? e : NEG_BIG;       // mask (exp underflows to exactly 0)
      const float p = __expf(e - Mr);    // p in [0,1]
      sl += p;
      ph[t] = (_Float16)p;
    }
    *(v4h*)&ptile[pb][rtp][mp][c0] = ph;
    __syncthreads();

    // ---- stage 2: 4x WMMA (no rescale needed) ----
    v16h A[2], Bm[2];
    for (int rt = 0; rt < 2; ++rt) {     // A from LDS (documented f16 A layout)
      const v8h lo = *(const v8h*)&ptile[pb][rt][l15][kb];
      const v8h hi = *(const v8h*)&ptile[pb][rt][l15][16 + kb];
      for (int i = 0; i < 8; ++i) { A[rt][i] = lo[i]; A[rt][8 + i] = hi[i]; }
    }
    {                                    // B from transposed h (contiguous b128)
      const v8h lo0 = *(const v8h*)(hA + j0 + kb);
      const v8h hi0 = *(const v8h*)(hA + j0 + 16 + kb);
      const v8h lo1 = *(const v8h*)(hB + j0 + kb);
      const v8h hi1 = *(const v8h*)(hB + j0 + 16 + kb);
      for (int i = 0; i < 8; ++i) {
        Bm[0][i] = lo0[i]; Bm[0][8 + i] = hi0[i];
        Bm[1][i] = lo1[i]; Bm[1][8 + i] = hi1[i];
      }
    }
    for (int rt = 0; rt < 2; ++rt) {
      acc[rt][0] = __builtin_amdgcn_wmma_f32_16x16x32_f16(false, A[rt], false, Bm[0],
                                                          (short)0, acc[rt][0], false, false);
      acc[rt][1] = __builtin_amdgcn_wmma_f32_16x16x32_f16(false, A[rt], false, Bm[1],
                                                          (short)0, acc[rt][1], false, false);
    }
    pb ^= 1;
  }

  // ---- final denominators: reduce over the 8 lanes owning each row ----
  {
    float s = sl;
    for (int off = 1; off <= 4; off <<= 1) s += __shfl_xor(s, off, 32);
    if ((tid & 7) == 0) l_s[r] = s;
  }
  __syncthreads();

  // ---- epilogue: divide by softmax denominator ----
  for (int rt = 0; rt < 2; ++rt) {
    for (int rr = 0; rr < 8; ++rr) {
      const int mrow = rt * 16 + rr + (half ? 8 : 0);
      const float l = l_s[mrow];
      const float inv = l > 0.f ? 1.0f / l : 0.f;
      out[(size_t)(row0 + mrow) * GAT_D + nA] = acc[rt][0][rr] * inv;
      out[(size_t)(row0 + mrow) * GAT_D + nB] = acc[rt][1][rr] * inv;
    }
  }
}

// ---------------------------------------------------------------------------
// Launch
// ---------------------------------------------------------------------------
extern "C" void kernel_launch(void* const* d_in, const int* in_sizes, int n_in,
                              void* d_out, int out_size, void* d_ws, size_t ws_size,
                              hipStream_t stream) {
  const float* X     = (const float*)d_in[0];   // node_features (N, D_IN)
  const int*   adj   = (const int*)d_in[1];     // adjacency (N, N)
  const float* W     = (const float*)d_in[2];   // (D_OUT, D_IN)
  const float* bW    = (const float*)d_in[3];   // (D_OUT,)
  const float* a_vec = (const float*)d_in[4];   // (2*D_OUT,)
  const float* b_a   = (const float*)d_in[5];   // (1,)
  float* out = (float*)d_out;

  // workspace: h_t (f16, D x N, 4 MB) | f1 | f2 | gmax | w1 | w2 | cb
  _Float16* h_t = (_Float16*)d_ws;
  float* f1   = (float*)((char*)d_ws + (size_t)GAT_N * GAT_D * sizeof(_Float16));
  float* f2   = f1 + GAT_N;
  float* gmax = f2 + GAT_N;
  float* w1   = gmax + 1;
  float* w2   = w1 + GAT_D;
  float* cb   = w2 + GAT_D;

  gat_w12<<<dim3(1), dim3(256), 0, stream>>>(W, bW, a_vec, w1, w2, cb);
  gat_f12<<<dim3(GAT_N / 256), dim3(256), 0, stream>>>(X, w1, w2, cb, f1, f2);
  gat_gmax<<<dim3(1), dim3(256), 0, stream>>>(f2, gmax);
  gat_phase1<<<dim3(GAT_N / 16, GAT_D / 16), dim3(32), 0, stream>>>(X, W, bW, h_t);
  gat_phase2<<<dim3(GAT_N / 32), dim3(256), 0, stream>>>(
      adj, f1, f2, b_a, gmax, h_t, out);
}
